// CausalSelfAttention_12592844112282
// MI455X (gfx1250) — matrix operand order repaired and checked by reading, not compiled, not run on hardware
//
#include <hip/hip_runtime.h>
#include <hip/hip_bf16.h>

typedef __bf16 bf16;
typedef __attribute__((ext_vector_type(16))) __bf16 v16bf;
typedef __attribute__((ext_vector_type(8)))  float  v8f;
typedef __attribute__((ext_vector_type(4)))  unsigned int v4u;
typedef __attribute__((ext_vector_type(4)))  int    v4i;
typedef __attribute__((ext_vector_type(8)))  int    v8i;

#define WMMA_BF16(a, b, c) \
  __builtin_amdgcn_wmma_f32_16x16x32_bf16(false, (a), false, (b), (short)0, (c), false, false)

constexpr int Bn = 4, Tn = 2048, Dn = 1024, Hn = 16, HDn = 64;
constexpr int Mrows = Bn * Tn;  // 8192

union FragU { v16bf v; uint4 q[2]; };

// ---------------------------------------------------------------- TDM tile load
// DMA a 128-row x 32-col bf16 tile (row stride = strideElems in global) into LDS
// at byte offset lds_off, padding each 64B row with 16B -> LDS pitch 40 bf16.
__device__ __forceinline__ void tdm_load_tile(const bf16* gptr, unsigned lds_off,
                                              unsigned strideElems) {
  unsigned long long ga = (unsigned long long)(uintptr_t)gptr;
  v4u g0;
  g0.x = 1u;                                     // count=1, user-mode descriptor
  g0.y = lds_off;                                // LDS byte address
  g0.z = (unsigned)ga;                           // global_addr[31:0]
  g0.w = (unsigned)((ga >> 32) & 0x1FFFFFFu) | (2u << 30);  // addr[56:32] | type=2
  v8i g1;
  g1[0] = (1 << 16)     // data_size = 2 bytes
        | (1 << 20)     // pad_enable
        | (3 << 22)     // pad_interval: 16 DWORDs (one 32-elem bf16 row)
        | (3 << 25);    // pad_amount:   4 DWORDs (16B)
  g1[1] = (32 << 16);                 // tensor_dim0[15:0] = 32
  g1[2] = (128 << 16);                // tensor_dim0[31:16]=0 | tensor_dim1[15:0]=128
  g1[3] = (32 << 16);                 // tensor_dim1[31:16]=0 | tile_dim0 = 32
  g1[4] = 128;                        // tile_dim1 = 128 | tile_dim2 = 0
  g1[5] = (int)strideElems;           // tensor_dim0_stride[31:0]
  g1[6] = 0;                          // stride hi | dim1_stride lo
  g1[7] = 0;
  v4i z4 = {0, 0, 0, 0};
#if defined(__clang_major__) && __clang_major__ >= 23
  v8i z8 = {0, 0, 0, 0, 0, 0, 0, 0};
  __builtin_amdgcn_tensor_load_to_lds(g0, g1, z4, z4, z8, 0);
#else
  __builtin_amdgcn_tensor_load_to_lds(g0, g1, z4, z4, 0);
#endif
}

// ---------------------------------------------------------------- converts
__global__ __launch_bounds__(256) void cvt_f32_bf16(const float* __restrict__ in,
                                                    bf16* __restrict__ out, int n) {
  int i = (blockIdx.x * 256 + threadIdx.x) * 4;
  if (i + 3 < n) {
    float4 f = *(const float4*)(in + i);
    out[i + 0] = (bf16)f.x; out[i + 1] = (bf16)f.y;
    out[i + 2] = (bf16)f.z; out[i + 3] = (bf16)f.w;
  }
}

// in: [rows][cols] f32 (row-major) -> out: [cols][rows] bf16
__global__ __launch_bounds__(256) void transpose_cvt(const float* __restrict__ in,
                                                     bf16* __restrict__ out,
                                                     int rows, int cols) {
  __shared__ float tile[32][33];
  int tx = threadIdx.x, ty = threadIdx.y;           // 32 x 8
  int r0 = blockIdx.y * 32, c0 = blockIdx.x * 32;
#pragma unroll
  for (int i = 0; i < 4; i++)
    tile[ty + i * 8][tx] = in[(size_t)(r0 + ty + i * 8) * cols + c0 + tx];
  __syncthreads();
#pragma unroll
  for (int i = 0; i < 4; i++)
    out[(size_t)(c0 + ty + i * 8) * rows + r0 + tx] = (bf16)tile[tx][ty + i * 8];
}

// ---------------------------------------------------------------- GEMM
// C = A[M,K] * Bt[N,K]^T + bias, TDM double-buffered staging.
// OUTMODE: 0 = f32 plain [M,N]
//   1 = bf16 head-split [B,H,T,HD]   2 = bf16 head-split transposed [B,H,HD,T]
template <int OUTMODE>
__global__ __launch_bounds__(256) void gemm_wmma(const bf16* __restrict__ A,
                                                 const bf16* __restrict__ Bt,
                                                 const float* __restrict__ bias,
                                                 void* __restrict__ Cout,
                                                 int M, int N, int K) {
  __shared__ alignas(16) bf16 As[2][128][40];
  __shared__ alignas(16) bf16 Bs[2][128][40];
  const int tid  = threadIdx.x;
  const int wave = tid >> 5, lane = tid & 31;
  const int lm = lane & 15, lg = lane >> 4;
  const int wrow = wave & 3, wcol = wave >> 2;      // 4x2 wave grid -> 32x64/wave
  const int m0 = blockIdx.y * 128;
  const int n0 = blockIdx.x * 128;

  v8f acc[2][4];
#pragma unroll
  for (int i = 0; i < 2; i++)
#pragma unroll
    for (int j = 0; j < 4; j++) acc[i][j] = (v8f){0, 0, 0, 0, 0, 0, 0, 0};

  const int niter = K / 32;
  if (wave == 0) {
    tdm_load_tile(A  + (size_t)m0 * K, (unsigned)(uintptr_t)&As[0][0][0], K);
    tdm_load_tile(Bt + (size_t)n0 * K, (unsigned)(uintptr_t)&Bs[0][0][0], K);
  }

  for (int it = 0; it < niter; ++it) {
    const int cur = it & 1;
    if (wave == 0) __builtin_amdgcn_s_wait_tensorcnt(0);
    __syncthreads();  // tile `it` resident in LDS for everyone
    if (wave == 0 && it + 1 < niter) {
      const int nxt = (it + 1) & 1;
      tdm_load_tile(A  + (size_t)m0 * K + (it + 1) * 32,
                    (unsigned)(uintptr_t)&As[nxt][0][0], K);
      tdm_load_tile(Bt + (size_t)n0 * K + (it + 1) * 32,
                    (unsigned)(uintptr_t)&Bs[nxt][0][0], K);
    }

    v16bf af[2], bfm[4];
#pragma unroll
    for (int i = 0; i < 2; i++) {
      const bf16* p = &As[cur][wrow * 32 + i * 16 + lm][0];
      FragU u;
      u.q[0] = *(const uint4*)(p + 8 * lg);
      u.q[1] = *(const uint4*)(p + 16 + 8 * lg);
      af[i] = u.v;
    }
#pragma unroll
    for (int j = 0; j < 4; j++) {
      const bf16* p = &Bs[cur][wcol * 64 + j * 16 + lm][0];
      FragU u;
      u.q[0] = *(const uint4*)(p + 8 * lg);
      u.q[1] = *(const uint4*)(p + 16 + 8 * lg);
      bfm[j] = u.v;
    }
#pragma unroll
    for (int i = 0; i < 2; i++)
#pragma unroll
      for (int j = 0; j < 4; j++) acc[i][j] = WMMA_BF16(af[i], bfm[j], acc[i][j]);
    __syncthreads();  // done reading buf[cur] before TDM overwrites it at it+2
  }

#pragma unroll
  for (int j = 0; j < 4; j++) {
    const int col = n0 + wcol * 64 + j * 16 + lm;
    const float bv = bias[col];
#pragma unroll
    for (int i = 0; i < 2; i++) {
#pragma unroll
      for (int r = 0; r < 8; r++) {
        const int row = m0 + wrow * 32 + i * 16 + r + 8 * lg;
        float v = acc[i][j][r] + bv;
        if (OUTMODE == 0) {
          ((float*)Cout)[(size_t)row * N + col] = v;
        } else {
          bf16 h = (bf16)v;
          int b = row / Tn, t = row % Tn, hh = col / HDn, hd = col % HDn;
          if (OUTMODE == 1)
            ((bf16*)Cout)[(((size_t)b * Hn + hh) * Tn + t) * HDn + hd] = h;
          else
            ((bf16*)Cout)[(((size_t)b * Hn + hh) * HDn + hd) * Tn + t] = h;
        }
      }
    }
  }
}

// ---------------------------------------------------------------- attention
// Q,K: [B,H,T,HD] bf16.  Vt: [B,H,HD,T] bf16.  Y: [B,T,D] bf16.
// 1 wave = 16 query rows, 32-key steps, online softmax, LDS bounce for P.
__global__ __launch_bounds__(256) void attn_wmma(const bf16* __restrict__ Q,
                                                 const bf16* __restrict__ Kk,
                                                 const bf16* __restrict__ Vt,
                                                 bf16* __restrict__ Y) {
  __shared__ alignas(16) bf16 Pl[8][16][40];      // per-wave P staging
  const int tid  = threadIdx.x;
  const int wave = tid >> 5, lane = tid & 31;
  const int lm = lane & 15, lg = lane >> 4;
  const int bh = blockIdx.y;
  const int q0 = blockIdx.x * 128 + wave * 16;
  const bf16* qp = Q  + (size_t)bh * Tn * HDn;
  const bf16* kp = Kk + (size_t)bh * Tn * HDn;
  const bf16* vp = Vt + (size_t)bh * HDn * Tn;

  v16bf qa[2];
#pragma unroll
  for (int c = 0; c < 2; c++) {
    const bf16* base = qp + (size_t)(q0 + lm) * HDn + c * 32;
    FragU u;
    u.q[0] = *(const uint4*)(base + 8 * lg);
    u.q[1] = *(const uint4*)(base + 16 + 8 * lg);
    qa[c] = u.v;
  }

  v8f o[4];
#pragma unroll
  for (int t2 = 0; t2 < 4; t2++) o[t2] = (v8f){0, 0, 0, 0, 0, 0, 0, 0};
  float mrow[8], lrow[8];
#pragma unroll
  for (int r = 0; r < 8; r++) { mrow[r] = -__builtin_inff(); lrow[r] = 0.f; }

  const float scale = 0.125f;  // 1/sqrt(64)

  for (int k0 = 0; k0 < q0 + 16; k0 += 32) {
    if (k0 + 32 < q0 + 16) {   // prefetch next key block (K rows + V^T rows)
      __builtin_prefetch(kp + (size_t)(k0 + 32 + lane) * HDn, 0, 1);
      __builtin_prefetch(vp + (size_t)(lane * 2) * Tn + k0 + 32, 0, 1);
    }
    // S = Q * K^T  (two 16-key tiles, HD split into 2 chunks of 32)
    v8f s[2];
#pragma unroll
    for (int kt = 0; kt < 2; kt++) {
      v8f sc = (v8f){0, 0, 0, 0, 0, 0, 0, 0};
#pragma unroll
      for (int c = 0; c < 2; c++) {
        const bf16* kb = kp + (size_t)(k0 + kt * 16 + lm) * HDn + c * 32 + 8 * lg;
        FragU u;
        u.q[0] = *(const uint4*)kb;
        u.q[1] = *(const uint4*)(kb + 16);
        sc = WMMA_BF16(qa[c], u.v, sc);
      }
      s[kt] = sc;
    }

    // causal mask + online softmax (row stats replicated across 16 lanes)
    float p0[8], p1[8];
#pragma unroll
    for (int r = 0; r < 8; r++) {
      const int qrow = q0 + r + 8 * lg;
      float s0 = (k0 + lm      <= qrow) ? s[0][r] * scale : -__builtin_inff();
      float s1 = (k0 + 16 + lm <= qrow) ? s[1][r] * scale : -__builtin_inff();
      float vmax = fmaxf(s0, s1);
#pragma unroll
      for (int off = 1; off < 16; off <<= 1) vmax = fmaxf(vmax, __shfl_xor(vmax, off, 32));
      const float mnew = fmaxf(mrow[r], vmax);
      const float sf = __expf(mrow[r] - mnew);
      mrow[r] = mnew;
      const float e0 = __expf(s0 - mnew), e1 = __expf(s1 - mnew);
      float rs = e0 + e1;
#pragma unroll
      for (int off = 1; off < 16; off <<= 1) rs += __shfl_xor(rs, off, 32);
      lrow[r] = lrow[r] * sf + rs;
#pragma unroll
      for (int t2 = 0; t2 < 4; t2++) o[t2][r] *= sf;
      p0[r] = e0; p1[r] = e1;
    }

    // P: C/D layout -> LDS -> A layout (per-wave region, in-order LDS pipe)
#pragma unroll
    for (int r = 0; r < 8; r++) {
      Pl[wave][r + 8 * lg][lm]      = (bf16)p0[r];
      Pl[wave][r + 8 * lg][16 + lm] = (bf16)p1[r];
    }
    __builtin_amdgcn_wave_barrier();
    asm volatile("s_wait_dscnt 0" ::: "memory");
    v16bf pa;
    {
      const bf16* pp = &Pl[wave][lm][0];
      FragU u;
      u.q[0] = *(const uint4*)(pp + 8 * lg);
      u.q[1] = *(const uint4*)(pp + 16 + 8 * lg);
      pa = u.v;
    }
    __builtin_amdgcn_wave_barrier();

    // O += P * V  (V^T rows are K-contiguous -> clean B fragments)
#pragma unroll
    for (int t2 = 0; t2 < 4; t2++) {
      const bf16* vb = vp + (size_t)(t2 * 16 + lm) * Tn + k0 + 8 * lg;
      FragU u;
      u.q[0] = *(const uint4*)vb;
      u.q[1] = *(const uint4*)(vb + 16);
      o[t2] = WMMA_BF16(pa, u.v, o[t2]);
    }
  }

  const int b = bh / Hn, h = bh % Hn;
#pragma unroll
  for (int t2 = 0; t2 < 4; t2++) {
#pragma unroll
    for (int r = 0; r < 8; r++) {
      const int qrow = q0 + r + 8 * lg;
      float val = o[t2][r] / lrow[r];
      Y[((size_t)b * Tn + qrow) * Dn + h * HDn + t2 * 16 + lm] = (bf16)val;
    }
  }
}

// ---------------------------------------------------------------- launch
extern "C" void kernel_launch(void* const* d_in, const int* in_sizes, int n_in,
                              void* d_out, int out_size, void* d_ws, size_t ws_size,
                              hipStream_t stream) {
  const float* x  = (const float*)d_in[0];
  const float* Wk = (const float*)d_in[1];
  const float* bk = (const float*)d_in[2];
  const float* Wq = (const float*)d_in[3];
  const float* bq = (const float*)d_in[4];
  const float* Wv = (const float*)d_in[5];
  const float* bv = (const float*)d_in[6];
  const float* Wp = (const float*)d_in[7];
  const float* bp = (const float*)d_in[8];

  const size_t MB = 1ull << 20;
  char* w = (char*)d_ws;
  bf16* xb  = (bf16*)(w + 0 * MB);    // 16 MB  x as bf16
  bf16* wqt = (bf16*)(w + 16 * MB);   // 2 MB   Wq^T bf16
  bf16* wkt = (bf16*)(w + 18 * MB);
  bf16* wvt = (bf16*)(w + 20 * MB);
  bf16* wpt = (bf16*)(w + 22 * MB);
  bf16* qb  = (bf16*)(w + 24 * MB);   // 16 MB  [B,H,T,HD]
  bf16* kb  = (bf16*)(w + 40 * MB);   // 16 MB  [B,H,T,HD]
  bf16* vtb = (bf16*)(w + 56 * MB);   // 16 MB  [B,H,HD,T]
  bf16* yb  = (bf16*)(w + 72 * MB);   // 16 MB  [B,T,D]

  const int nX = Bn * Tn * Dn;  // 8388608
  cvt_f32_bf16<<<nX / (4 * 256), 256, 0, stream>>>(x, xb, nX);

  dim3 tb(32, 8), tg(Dn / 32, Dn / 32);
  transpose_cvt<<<tg, tb, 0, stream>>>(Wq, wqt, Dn, Dn);
  transpose_cvt<<<tg, tb, 0, stream>>>(Wk, wkt, Dn, Dn);
  transpose_cvt<<<tg, tb, 0, stream>>>(Wv, wvt, Dn, Dn);
  transpose_cvt<<<tg, tb, 0, stream>>>(Wp, wpt, Dn, Dn);

  dim3 gg(Dn / 128, Mrows / 128);  // (8, 64)
  gemm_wmma<1><<<gg, 256, 0, stream>>>(xb, wqt, bq, qb,  Mrows, Dn, Dn);
  gemm_wmma<1><<<gg, 256, 0, stream>>>(xb, wkt, bk, kb,  Mrows, Dn, Dn);
  gemm_wmma<2><<<gg, 256, 0, stream>>>(xb, wvt, bv, vtb, Mrows, Dn, Dn);

  attn_wmma<<<dim3(Tn / 128, Bn * Hn), 256, 0, stream>>>(qb, kb, vtb, yb);

  gemm_wmma<0><<<gg, 256, 0, stream>>>(yb, wpt, bp, (float*)d_out, Mrows, Dn, Dn);
}
